// ProbabilisticMultiHeadAttention_21165598835351
// MI455X (gfx1250) — compile-verified, matrix operand order checked by reading, not run on hardware
//
#include <hip/hip_runtime.h>
#include <hip/hip_bf16.h>

// ---------------------------------------------------------------------------
// Probabilistic MHA forward for MI455X (gfx1250), wave32 + WMMA bf16 pipeline.
//   qkv_mu = mu    @ Wqkv_mu + b   -> bf16 workspace
//   qkv_sg = sigma @ Wqkv_sg + b   -> bf16 workspace
//   flash attention: S = (Qm Km^T + alpha Qs Ks^T)/8, causal softmax,
//                    Om = P Vm, Os = P Vs               -> bf16 workspace
//   out_mu = Om @ Wproj_mu + b ; out_sg = Os @ Wproj_sg + b -> d_out (fp32)
// Uses v_wmma_f32_16x16x32_bf16 + global_load_async_to_lds_b128 (ASYNCcnt)
// for the straight-copy bf16 tiles, global_prefetch for weight streaming.
// ---------------------------------------------------------------------------

#define D_MODEL 1024
#define NHEADS  16
#define DK      64
#define BSZ     2
#define SEQL    2048
#define MROWS   (BSZ * SEQL)   // 4096
#define QKV_N   (3 * D_MODEL)  // 3072

typedef __bf16 bf16;
typedef __attribute__((ext_vector_type(16))) __bf16 v16bf;
typedef __attribute__((ext_vector_type(8)))  float  v8f;

// ---- gfx1250 helpers -------------------------------------------------------

__device__ __forceinline__ v8f wmma_bf16(v16bf a, v16bf b, v8f c) {
  // D = A(16x32) * B(32x16) + C, fp32 accumulate
  return __builtin_amdgcn_wmma_f32_16x16x32_bf16(
      /*neg_a=*/false, a, /*neg_b=*/false, b,
      /*c_mod=*/(short)0, c, /*reuse_a=*/false, /*reuse_b=*/false);
}

// LDS byte offset of a __shared__ object (addrspace(3) offset).
__device__ __forceinline__ unsigned lds_off(const void* p) {
  return (unsigned)(size_t)((__attribute__((address_space(3))) const void*)p);
}

// Async global -> LDS copy, 16 bytes per lane, GV addressing, ASYNCcnt.
__device__ __forceinline__ void async_ld_b128(unsigned ldsByteOff, const void* g) {
  asm volatile("global_load_async_to_lds_b128 %0, %1, off"
               :: "v"(ldsByteOff), "v"((unsigned long long)(size_t)g)
               : "memory");
}
__device__ __forceinline__ void wait_async0() {
  asm volatile("s_wait_asynccnt 0" ::: "memory");
}

// Load a 16x32 bf16 operand fragment from an LDS tile laid out row-major
// ([16 rows][stride bf16]).  Per ISA 7.12.2 (16-bit A-matrix 16x32):
//   lane = {g,r}: r = lane&15 selects row, g = lane>>4 selects K-half;
//   VGPR v<4  holds K = 2v   + 8g .. +1
//   VGPR v>=4 holds K = 16+2(v-4) + 8g .. +1
// B operands use the symmetric layout (lane&15 = N column) from a [N][K] tile.
__device__ __forceinline__ v16bf load_frag(const bf16* tile, int stride, int lane) {
  const int r = lane & 15;
  const int g = (lane >> 4) & 1;
  const bf16* p = tile + r * stride + g * 8;
  v16bf f;
#pragma unroll
  for (int v = 0; v < 8; ++v) {
    const int kb = (v < 4) ? (v * 2) : (16 + (v - 4) * 2);
    f[2 * v]     = p[kb];
    f[2 * v + 1] = p[kb + 1];
  }
  return f;
}

// ---- Generic 64x64 block-tile GEMM:  C[M,N] = A[M,K] * W[K,N] + bias -------
// 128 threads = 4 waves, wave = 32x32 quadrant (2x2 WMMA tiles), BK = 64
// (8 WMMAs per wave per barrier pair).

template <bool A_IS_BF16, bool OUT_IS_BF16>
__global__ __launch_bounds__(128)
void gemm64_bias(const void* __restrict__ Aptr,
                 const float* __restrict__ W,
                 const float* __restrict__ bias,
                 void* __restrict__ Cptr,
                 int M, int N, int K) {
  __shared__ bf16 sA[64 * 72];  // [64 rows][64 K] + pad
  __shared__ bf16 sB[64 * 72];  // [64 cols][64 K] + pad (W transposed)

  const int n0   = blockIdx.x * 64;
  const int m0   = blockIdx.y * 64;
  const int tid  = threadIdx.x;
  const int lane = tid & 31;
  const int wid  = tid >> 5;
  const int wm   = (wid >> 1) * 32;
  const int wn   = (wid & 1) * 32;

  const float* Af = (const float*)Aptr;
  const bf16*  Ab = (const bf16*)Aptr;
  const unsigned sAoff = lds_off(sA);

  v8f acc[2][2];
#pragma unroll
  for (int mt = 0; mt < 2; ++mt)
#pragma unroll
    for (int nt = 0; nt < 2; ++nt)
#pragma unroll
      for (int r = 0; r < 8; ++r) acc[mt][nt][r] = 0.0f;

  for (int k0 = 0; k0 < K; k0 += 64) {
    __syncthreads();
    // ---- Stage A tile (64x64 bf16, row-major) ----
    if (A_IS_BF16) {
      // Straight bf16 copy: async global->LDS, 16B per transfer (ASYNCcnt).
      for (int i = tid; i < 512; i += 128) {
        const int r = i >> 3, seg = i & 7;
        async_ld_b128(sAoff + r * 144 + seg * 16,
                      Ab + (long)(m0 + r) * K + k0 + seg * 8);
      }
    } else {
#pragma unroll 4
      for (int i = tid; i < 64 * 64; i += 128) {
        const int r = i >> 6, c = i & 63;
        sA[r * 72 + c] = (bf16)Af[(long)(m0 + r) * K + (k0 + c)];
      }
    }
    // ---- Stage W tile transposed -> sB[n][k] (coalesced reads along n) ----
#pragma unroll 4
    for (int i = tid; i < 64 * 64; i += 128) {
      const int k = i >> 6, n = i & 63;
      sB[n * 72 + k] = (bf16)W[(long)(k0 + k) * N + (n0 + n)];
    }
    // Prefetch next W k-tile while we compute on this one.
    if (k0 + 64 < K) {
      const int pr = tid & 63, ph = (tid >> 6) & 1;
      __builtin_prefetch(W + (long)(k0 + 64 + pr) * N + n0 + ph * 32, 0, 1);
    }
    if (A_IS_BF16) wait_async0();
    __syncthreads();

#pragma unroll
    for (int kk = 0; kk < 64; kk += 32) {
      v16bf afrag[2], bfrag[2];
#pragma unroll
      for (int t = 0; t < 2; ++t) {
        afrag[t] = load_frag(sA + (wm + t * 16) * 72 + kk, 72, lane);
        bfrag[t] = load_frag(sB + (wn + t * 16) * 72 + kk, 72, lane);
      }
#pragma unroll
      for (int mt = 0; mt < 2; ++mt)
#pragma unroll
        for (int nt = 0; nt < 2; ++nt)
          acc[mt][nt] = wmma_bf16(afrag[mt], bfrag[nt], acc[mt][nt]);
    }
  }

  // C layout (16x16 f32): lane&15 = N col, VGPR r -> row r + 8*(lane>>4).
  const int nlane = lane & 15;
  const int moff  = (lane >> 4) * 8;
#pragma unroll
  for (int mt = 0; mt < 2; ++mt)
#pragma unroll
    for (int nt = 0; nt < 2; ++nt) {
      const int col = n0 + wn + nt * 16 + nlane;
      const float bv = bias[col];
#pragma unroll
      for (int r = 0; r < 8; ++r) {
        const long row = m0 + wm + mt * 16 + moff + r;
        const float v = acc[mt][nt][r] + bv;
        if (OUT_IS_BF16) ((bf16*)Cptr)[row * N + col] = (bf16)v;
        else             ((float*)Cptr)[row * N + col] = v;
      }
    }
}

// ---- Fused causal flash attention (dual mu/sigma scores, shared softmax) ---
// grid = (SEQL/64, BSZ*NHEADS), block = 128 (4 waves); wave w owns 16 q-rows.

__global__ __launch_bounds__(128)
void prob_attn(const bf16* __restrict__ qkv_mu,
               const bf16* __restrict__ qkv_sg,
               const float* __restrict__ alphaPtr,
               bf16* __restrict__ attn_mu,
               bf16* __restrict__ attn_sg) {
  __shared__ bf16 sQm[64 * 72];
  __shared__ bf16 sQs[64 * 72];
  __shared__ bf16 sKm[64 * 72];      // [key][d]   (already B-layout for QK^T)
  __shared__ bf16 sKs[64 * 72];
  __shared__ bf16 sVm[64 * 72];      // [d][key]   (B-layout for P*V)
  __shared__ bf16 sVs[64 * 72];
  __shared__ bf16 sP[4][16 * 72];    // per-wave P scratch: acc-layout -> A-layout

  const int q0   = blockIdx.x * 64;
  const int bh   = blockIdx.y;
  const int b    = bh / NHEADS;
  const int h    = bh % NHEADS;
  const int tid  = threadIdx.x;
  const int lane = tid & 31;
  const int wave = tid >> 5;

  const float alpha = alphaPtr[0];
  const float scale = 0.125f;  // 1/sqrt(64)

  const long rowbase = (long)b * SEQL;
  const bf16* Qm = qkv_mu + rowbase * QKV_N + h * DK;
  const bf16* Km = qkv_mu + rowbase * QKV_N + D_MODEL + h * DK;
  const bf16* Vm = qkv_mu + rowbase * QKV_N + 2 * D_MODEL + h * DK;
  const bf16* Qs = qkv_sg + rowbase * QKV_N + h * DK;
  const bf16* Ks = qkv_sg + rowbase * QKV_N + D_MODEL + h * DK;
  const bf16* Vs = qkv_sg + rowbase * QKV_N + 2 * D_MODEL + h * DK;

  const unsigned oQm = lds_off(sQm), oQs = lds_off(sQs);
  const unsigned oKm = lds_off(sKm), oKs = lds_off(sKs);

  // Stage the 64x64 Q tiles once via async global->LDS (straight bf16 copy).
  for (int i = tid; i < 512; i += 128) {
    const int r = i >> 3, seg = i & 7;
    const unsigned l = r * 144 + seg * 16;
    const long gr = (long)(q0 + r) * QKV_N + seg * 8;
    async_ld_b128(oQm + l, Qm + gr);
    async_ld_b128(oQs + l, Qs + gr);
  }
  wait_async0();
  __syncthreads();
  const bf16* qtm = sQm + (wave * 16) * 72;
  const bf16* qts = sQs + (wave * 16) * 72;
  const v16bf qm0 = load_frag(qtm, 72, lane);
  const v16bf qm1 = load_frag(qtm + 32, 72, lane);
  const v16bf qs0 = load_frag(qts, 72, lane);
  const v16bf qs1 = load_frag(qts + 32, 72, lane);

  v8f Om[4], Os[4];
#pragma unroll
  for (int dt = 0; dt < 4; ++dt)
#pragma unroll
    for (int r = 0; r < 8; ++r) { Om[dt][r] = 0.0f; Os[dt][r] = 0.0f; }

  float row_m[8], row_l[8];
#pragma unroll
  for (int r = 0; r < 8; ++r) { row_m[r] = -3.0e30f; row_l[r] = 0.0f; }

  const int nlane    = lane & 15;
  const int moff     = (lane >> 4) * 8;
  const int qrowBase = q0 + wave * 16 + moff;

  for (int kt0 = 0; kt0 <= q0; kt0 += 64) {
    __syncthreads();
    // K tiles: straight copy -> async global->LDS.
    for (int i = tid; i < 512; i += 128) {
      const int r = i >> 3, seg = i & 7;
      const unsigned l = r * 144 + seg * 16;
      const long gr = (long)(kt0 + r) * QKV_N + seg * 8;
      async_ld_b128(oKm + l, Km + gr);
      async_ld_b128(oKs + l, Ks + gr);
    }
    // V tiles need a transpose -> manual staging ([d][key]).
    for (int i = tid; i < 64 * 64; i += 128) {
      const int r = i >> 6, c = i & 63;
      const long gr = (long)(kt0 + r) * QKV_N;
      sVm[c * 72 + r] = Vm[gr + c];
      sVs[c * 72 + r] = Vs[gr + c];
    }
    wait_async0();
    __syncthreads();

    // Scores: 16 q-rows x 64 keys, mu + alpha*sigma, scaled.
    float s[4][8];
#pragma unroll
    for (int nt = 0; nt < 4; ++nt) {
      const bf16* ktm = sKm + (nt * 16) * 72;
      const bf16* kts = sKs + (nt * 16) * 72;
      v8f am, as;
#pragma unroll
      for (int r = 0; r < 8; ++r) { am[r] = 0.0f; as[r] = 0.0f; }
      am = wmma_bf16(qm0, load_frag(ktm, 72, lane), am);
      am = wmma_bf16(qm1, load_frag(ktm + 32, 72, lane), am);
      as = wmma_bf16(qs0, load_frag(kts, 72, lane), as);
      as = wmma_bf16(qs1, load_frag(kts + 32, 72, lane), as);
#pragma unroll
      for (int r = 0; r < 8; ++r) s[nt][r] = (am[r] + alpha * as[r]) * scale;
    }

    if (kt0 == q0) {  // only the diagonal tile needs causal masking
#pragma unroll
      for (int nt = 0; nt < 4; ++nt) {
        const int kcol = kt0 + nt * 16 + nlane;
#pragma unroll
        for (int r = 0; r < 8; ++r)
          if (kcol > qrowBase + r) s[nt][r] = -3.0e30f;
      }
    }

    // Online softmax: row stats reduced across the 16-lane half-wave.
    float mnew[8], corr[8], psum[8];
#pragma unroll
    for (int r = 0; r < 8; ++r) {
      float mx = row_m[r];
#pragma unroll
      for (int nt = 0; nt < 4; ++nt) mx = fmaxf(mx, s[nt][r]);
#pragma unroll
      for (int off = 1; off < 16; off <<= 1)
        mx = fmaxf(mx, __shfl_xor(mx, off, 16));
      mnew[r]  = mx;
      corr[r]  = __expf(row_m[r] - mx);
      row_m[r] = mx;
      psum[r]  = 0.0f;
    }

    // P = exp(s - m): accumulate row sums and spill to per-wave LDS patch.
    bf16* pW = sP[wave];
#pragma unroll
    for (int nt = 0; nt < 4; ++nt)
#pragma unroll
      for (int r = 0; r < 8; ++r) {
        const float p = __expf(s[nt][r] - mnew[r]);
        psum[r] += p;
        pW[(moff + r) * 72 + nt * 16 + nlane] = (bf16)p;
      }
#pragma unroll
    for (int r = 0; r < 8; ++r) {
      float ps = psum[r];
#pragma unroll
      for (int off = 1; off < 16; off <<= 1) ps += __shfl_xor(ps, off, 16);
      row_l[r] = row_l[r] * corr[r] + ps;
    }

    // Rescale running outputs, then O += P @ V (per-wave LDS, in-order ds).
#pragma unroll
    for (int dt = 0; dt < 4; ++dt)
#pragma unroll
      for (int r = 0; r < 8; ++r) { Om[dt][r] *= corr[r]; Os[dt][r] *= corr[r]; }

    const v16bf p0 = load_frag(pW, 72, lane);
    const v16bf p1 = load_frag(pW + 32, 72, lane);
#pragma unroll
    for (int dt = 0; dt < 4; ++dt) {
      const bf16* vtm = sVm + (dt * 16) * 72;
      const bf16* vts = sVs + (dt * 16) * 72;
      Om[dt] = wmma_bf16(p0, load_frag(vtm, 72, lane), Om[dt]);
      Om[dt] = wmma_bf16(p1, load_frag(vtm + 32, 72, lane), Om[dt]);
      Os[dt] = wmma_bf16(p0, load_frag(vts, 72, lane), Os[dt]);
      Os[dt] = wmma_bf16(p1, load_frag(vts + 32, 72, lane), Os[dt]);
    }
  }

  // Normalize and write [b, q, h*dk] as bf16 for the projection GEMM.
#pragma unroll
  for (int r = 0; r < 8; ++r) {
    const float inv = 1.0f / row_l[r];
    const long grow = rowbase + qrowBase + r;
#pragma unroll
    for (int dt = 0; dt < 4; ++dt) {
      const int col = h * DK + dt * 16 + nlane;
      attn_mu[grow * D_MODEL + col] = (bf16)(Om[dt][r] * inv);
      attn_sg[grow * D_MODEL + col] = (bf16)(Os[dt][r] * inv);
    }
  }
}

// ---- Host-side launch ------------------------------------------------------

extern "C" void kernel_launch(void* const* d_in, const int* in_sizes, int n_in,
                              void* d_out, int out_size, void* d_ws, size_t ws_size,
                              hipStream_t stream) {
  const float* mu       = (const float*)d_in[0];
  const float* sigma    = (const float*)d_in[1];
  const float* Wqkv_mu  = (const float*)d_in[2];
  const float* bqkv_mu  = (const float*)d_in[3];
  const float* Wqkv_sg  = (const float*)d_in[4];
  const float* bqkv_sg  = (const float*)d_in[5];
  const float* Wproj_mu = (const float*)d_in[6];
  const float* bproj_mu = (const float*)d_in[7];
  const float* Wproj_sg = (const float*)d_in[8];
  const float* bproj_sg = (const float*)d_in[9];
  const float* alpha    = (const float*)d_in[10];

  char* ws = (char*)d_ws;
  const size_t qkvBytes  = (size_t)MROWS * QKV_N * sizeof(bf16);   // 24 MB
  const size_t attnBytes = (size_t)MROWS * D_MODEL * sizeof(bf16); // 8 MB
  bf16* qkv_mu  = (bf16*)(ws);
  bf16* qkv_sg  = (bf16*)(ws + qkvBytes);
  bf16* attn_mu = (bf16*)(ws + 2 * qkvBytes);
  bf16* attn_sg = (bf16*)(ws + 2 * qkvBytes + attnBytes);

  float* out_mu = (float*)d_out;
  float* out_sg = out_mu + (size_t)MROWS * D_MODEL;

  const dim3 blk(128);

  // QKV projections (fp32 in, bf16 out).
  const dim3 gq(QKV_N / 64, MROWS / 64);
  gemm64_bias<false, true><<<gq, blk, 0, stream>>>(mu,    Wqkv_mu, bqkv_mu, qkv_mu, MROWS, QKV_N, D_MODEL);
  gemm64_bias<false, true><<<gq, blk, 0, stream>>>(sigma, Wqkv_sg, bqkv_sg, qkv_sg, MROWS, QKV_N, D_MODEL);

  // Fused causal attention.
  const dim3 ga(SEQL / 64, BSZ * NHEADS);
  prob_attn<<<ga, blk, 0, stream>>>(qkv_mu, qkv_sg, alpha, attn_mu, attn_sg);

  // Output projections (bf16 in, fp32 out straight into d_out).
  const dim3 gp(D_MODEL / 64, MROWS / 64);
  gemm64_bias<true, false><<<gp, blk, 0, stream>>>(attn_mu, Wproj_mu, bproj_mu, out_mu, MROWS, D_MODEL, D_MODEL);
  gemm64_bias<true, false><<<gp, blk, 0, stream>>>(attn_sg, Wproj_sg, bproj_sg, out_sg, MROWS, D_MODEL, D_MODEL);
}
